// CausalSelfAttention_59923383714029
// MI455X (gfx1250) — matrix operand order repaired and checked by reading, compiled, not run on hardware
//
#include <hip/hip_runtime.h>
#include <stdint.h>

typedef _Float16 f16;
typedef __attribute__((ext_vector_type(16))) _Float16 v16h;
typedef __attribute__((ext_vector_type(8)))  float    v8f;
typedef uint32_t u32x4 __attribute__((ext_vector_type(4)));
typedef int      i32x4 __attribute__((ext_vector_type(4)));
typedef int      i32x8 __attribute__((ext_vector_type(8)));

#define WMMA_F32_F16(a, b, c) \
  __builtin_amdgcn_wmma_f32_16x16x32_f16(false, (a), false, (b), (short)0, (c), false, false)

#if __has_builtin(__builtin_amdgcn_tensor_load_to_lds) && \
    __has_builtin(__builtin_amdgcn_s_wait_tensorcnt)
#define HAVE_TDM 1
#if __has_include(<hip/amd_detail/amd_gfx1250_TDM.h>)
#define TDM_ARITY6 1
#endif
#endif

// ---------------------------------------------------------------------------
// Fragment loaders: contiguous k order, shared by A and B (not the ISA table order).
// A (16x32 f16, MxK): lane M = lane%16, half = lane/16.
//   VGPR i : K = 2i + 16*half (contiguous, matches B loaders)
// ---------------------------------------------------------------------------
__device__ __forceinline__ v16h load_a_frag(const f16* __restrict__ base, int ld, int lane) {
  int M = lane & 15, hh = lane >> 4;
  const f16* r = base + (size_t)M * ld;
  v16h a;
#pragma unroll
  for (int i = 0; i < 8; ++i) {
    int kb = 2 * i + 16 * hh;
    a[2 * i]     = r[kb];
    a[2 * i + 1] = r[kb + 1];
  }
  return a;
}

// B (32x16 f16, KxN), row-major source B[k][N]: lane N = lane%16, half picks K 16-block.
__device__ __forceinline__ v16h load_b_frag(const f16* __restrict__ base, int ld, int lane) {
  int N = lane & 15, hh = lane >> 4;
  v16h b;
#pragma unroll
  for (int j = 0; j < 8; ++j) {
    int k = 2 * j + 16 * hh;
    b[2 * j]     = base[(size_t)k * ld + N];
    b[2 * j + 1] = base[(size_t)(k + 1) * ld + N];
  }
  return b;
}

// B fragment gathered from a transposed source: B[k][N] = src[N][k]  (for Q*K^T)
__device__ __forceinline__ v16h load_bt_frag(const f16* __restrict__ base, int ld, int lane) {
  int N = lane & 15, hh = lane >> 4;
  const f16* r = base + (size_t)N * ld;
  v16h b;
#pragma unroll
  for (int j = 0; j < 8; ++j) {
    int k = 2 * j + 16 * hh;
    b[2 * j]     = r[k];
    b[2 * j + 1] = r[k + 1];
  }
  return b;
}

// ---------------------------------------------------------------------------
// GEMM + bias:  C[M,N] = A[M,K] * B[K,N] + bias[N]
// Block = 256 threads (8 waves), tile 128x128, K-step 32.
// Waves arranged 2(M) x 4(N); wave tile 64x32 -> 8 WMMA / 6 frag loads per step.
// ---------------------------------------------------------------------------
template <typename TA, typename TOUT>
__global__ __launch_bounds__(256) void gemm_bias_wmma(
    const TA* __restrict__ A, const float* __restrict__ B,
    const float* __restrict__ bias, TOUT* __restrict__ C,
    int M, int N, int K) {
  constexpr int BM = 128, BN = 128, BK = 32;
  constexpr int ALD = BK + 2;  // 34 halves: 17-bank row stride, coprime with 64
  constexpr int BLD = BN + 2;  // 130
  __shared__ f16 As[BM * ALD];
  __shared__ f16 Bs[BK * BLD];

  int tid  = threadIdx.x;
  int lane = tid & 31, wid = tid >> 5;
  int wm = wid & 1, wn = wid >> 1;
  int row0 = blockIdx.y * BM;
  int col0 = blockIdx.x * BN;

  v8f z = {};
  v8f acc[4][2] = {{z, z}, {z, z}, {z, z}, {z, z}};

  for (int k0 = 0; k0 < K; k0 += BK) {
    for (int i = tid; i < BM * BK; i += 256) {
      int r = i >> 5, c = i & 31;
      As[r * ALD + c] = (f16)A[(size_t)(row0 + r) * K + k0 + c];
    }
    for (int i = tid; i < BK * BN; i += 256) {
      int r = i >> 7, c = i & 127;
      Bs[r * BLD + c] = (f16)B[(size_t)(k0 + r) * N + col0 + c];
    }
    if (k0 + BK < K) {  // prefetch next K tile (global_prefetch_b8)
      __builtin_prefetch(&A[(size_t)(row0 + (tid >> 1)) * K + k0 + BK + (tid & 1) * 16], 0, 3);
      __builtin_prefetch(&B[(size_t)(k0 + BK + (tid >> 3)) * N + col0 + (tid & 7) * 16], 0, 3);
    }
    __syncthreads();

    v16h af[4], bf[2];
#pragma unroll
    for (int mi = 0; mi < 4; ++mi)
      af[mi] = load_a_frag(&As[(wm * 64 + mi * 16) * ALD], ALD, lane);
#pragma unroll
    for (int j = 0; j < 2; ++j)
      bf[j] = load_b_frag(&Bs[wn * 32 + j * 16], BLD, lane);
#pragma unroll
    for (int mi = 0; mi < 4; ++mi)
#pragma unroll
      for (int j = 0; j < 2; ++j)
        acc[mi][j] = WMMA_F32_F16(af[mi], bf[j], acc[mi][j]);
    __syncthreads();
  }

  int colk = lane & 15, hh = lane >> 4;
#pragma unroll
  for (int mi = 0; mi < 4; ++mi)
#pragma unroll
    for (int j = 0; j < 2; ++j) {
      int cg = col0 + wn * 32 + j * 16 + colk;
      float bv = bias[cg];
#pragma unroll
      for (int r = 0; r < 8; ++r) {
        int rg = row0 + wm * 64 + mi * 16 + r + 8 * hh;
        C[(size_t)rg * N + cg] = (TOUT)(acc[mi][j][r] + bv);
      }
    }
}

// ---------------------------------------------------------------------------
// TDM issue: 2D tile [rows=32 x cols=64 f16] from global (row stride 3072 elem)
// into LDS with 8-half padding per row (-> 72-half LDS row stride).
// ---------------------------------------------------------------------------
#ifdef HAVE_TDM
__device__ __forceinline__ void tdm_load_tile_2d(const f16* gptr, void* lds_dst) {
  uint64_t ga   = (uint64_t)(uintptr_t)gptr;
  uint32_t ldsa = (uint32_t)(uintptr_t)lds_dst;
  u32x4 g0 = { 1u,                                   // count=1, valid user D#
               ldsa,                                 // lds_addr
               (uint32_t)ga,                         // global_addr[31:0]
               (uint32_t)((ga >> 32) & 0x1FFFFFFu) | (2u << 30) };  // ga[56:32] | type=2
  // group1: data_size=2B(1), pad_enable, pad_interval=32dw(4), pad_amount=4dw(3)
  uint32_t d0 = (1u << 16) | (1u << 20) | (4u << 22) | (3u << 25);
  uint32_t d1 = (64u << 16);        // tensor_dim0 = 64 (low 16 bits at [31:16])
  uint32_t d2 = (32u << 16);        // tensor_dim0 hi16 = 0 ; tensor_dim1 = 32
  uint32_t d3 = (64u << 16);        // tensor_dim1 hi16 = 0 ; tile_dim0 = 64
  uint32_t d4 = 32u;                // tile_dim1 = 32 ; tile_dim2 = 0
  uint32_t d5 = 3072u;              // tensor_dim0_stride low32
  i32x8 g1 = { (int)d0, (int)d1, (int)d2, (int)d3, (int)d4, (int)d5, 0, 0 };
  i32x4 gz4 = { 0, 0, 0, 0 };
#ifdef TDM_ARITY6
  i32x8 gz8 = { 0, 0, 0, 0, 0, 0, 0, 0 };
  __builtin_amdgcn_tensor_load_to_lds(g0, g1, gz4, gz4, gz8, 0);
#else
  __builtin_amdgcn_tensor_load_to_lds(g0, g1, gz4, gz4, 0);
#endif
}
#endif

// ---------------------------------------------------------------------------
// Flash attention (causal). qkv: f16 [B*T, 3072] (Q|K|V). y: f16 [B*T, 1024].
// Block: 8 waves, each owns a 16-row Q tile (128 q rows / block / head).
// K,V tiles staged by the Tensor Data Mover (wave 0), shared by all waves.
// ---------------------------------------------------------------------------
__global__ __launch_bounds__(256) void attn_wmma(const f16* __restrict__ qkv,
                                                 f16* __restrict__ y) {
  constexpr int T = 2048, C3 = 3072, DH = 64;
  constexpr int KLD = DH + 8;   // 72 halves; matches TDM pad (128B data + 16B pad / row)
  constexpr int PLD = 34;
  __shared__ f16 kt_lds[32 * KLD];
  __shared__ f16 vt_lds[32 * KLD];
  __shared__ f16 p_lds[8 * 16 * PLD];

  int tid = threadIdx.x, lane = tid & 31, wid = tid >> 5;
  int bh = blockIdx.y;
  int b = bh >> 4, head = bh & 15;
  int qb = blockIdx.x * 128;
  int q0 = qb + wid * 16;
  int colk = lane & 15, hh = lane >> 4;
  const float scale = 0.125f;  // 1/sqrt(64)

  // Q fragments: 16 rows x K=64 split into two k=32 chunks (held for entire loop)
  const f16* qbase = qkv + (size_t)(b * T + q0) * C3 + head * DH;
  v16h qa0 = load_a_frag(qbase,      C3, lane);
  v16h qa1 = load_a_frag(qbase + 32, C3, lane);

  v8f z = {};
  v8f o0 = z, o1 = z, o2 = z, o3 = z;
  float m[8], l[8];
#pragma unroll
  for (int r = 0; r < 8; ++r) { m[r] = -1e30f; l[r] = 0.0f; }

  f16* pw = p_lds + wid * 16 * PLD;
  int kt_end = qb + 128;  // uniform per block -> uniform barriers

  for (int kt = 0; kt < kt_end; kt += 32) {
    __syncthreads();  // K/V LDS safe to overwrite: all waves finished previous PV
#ifdef HAVE_TDM
    if (wid == 0) {   // wave-uniform; TDM ignores EXEC. One wave issues both DMAs.
      const f16* kg = qkv + (size_t)(b * T + kt) * C3 + 1024 + head * DH;
      tdm_load_tile_2d(kg,        kt_lds);
      tdm_load_tile_2d(kg + 1024, vt_lds);
      __builtin_amdgcn_s_wait_tensorcnt(0);
    }
#else
    for (int i = tid; i < 32 * DH; i += 256) {
      int r = i >> 6, c = i & 63;
      size_t g = (size_t)(b * T + kt + r) * C3 + 1024 + head * DH + c;
      kt_lds[r * KLD + c] = qkv[g];
      vt_lds[r * KLD + c] = qkv[g + 1024];
    }
#endif
    __syncthreads();

    // S = Q * K^T  (two 16-key score tiles, K-dim = 64 = 2 WMMA each)
    v16h kb00 = load_bt_frag(&kt_lds[0],             KLD, lane);
    v16h kb01 = load_bt_frag(&kt_lds[32],            KLD, lane);
    v16h kb10 = load_bt_frag(&kt_lds[16 * KLD],      KLD, lane);
    v16h kb11 = load_bt_frag(&kt_lds[16 * KLD + 32], KLD, lane);
    v8f s0 = z, s1 = z;
    s0 = WMMA_F32_F16(qa0, kb00, s0);
    s0 = WMMA_F32_F16(qa1, kb01, s0);
    s1 = WMMA_F32_F16(qa0, kb10, s1);
    s1 = WMMA_F32_F16(qa1, kb11, s1);

    // online softmax (rows r+8*hh, columns across 16-lane half-groups)
    float p0[8], p1[8], tm[8], ts[8];
#pragma unroll
    for (int r = 0; r < 8; ++r) {
      int qrow = q0 + r + 8 * hh;
      float v0 = (kt + colk      <= qrow) ? s0[r] * scale : -1e30f;
      float v1 = (kt + 16 + colk <= qrow) ? s1[r] * scale : -1e30f;
      p0[r] = v0; p1[r] = v1;
      tm[r] = fmaxf(v0, v1);
    }
#pragma unroll
    for (int mk = 1; mk <= 8; mk <<= 1)
#pragma unroll
      for (int r = 0; r < 8; ++r)
        tm[r] = fmaxf(tm[r], __shfl_xor(tm[r], mk, 32));
#pragma unroll
    for (int r = 0; r < 8; ++r) {
      float nm   = fmaxf(m[r], tm[r]);
      float corr = __expf(m[r] - nm);
      m[r] = nm;
      p0[r] = __expf(p0[r] - nm);
      p1[r] = __expf(p1[r] - nm);
      ts[r] = p0[r] + p1[r];
      l[r] *= corr;
      o0[r] *= corr; o1[r] *= corr; o2[r] *= corr; o3[r] *= corr;
    }
#pragma unroll
    for (int mk = 1; mk <= 8; mk <<= 1)
#pragma unroll
      for (int r = 0; r < 8; ++r)
        ts[r] += __shfl_xor(ts[r], mk, 32);
#pragma unroll
    for (int r = 0; r < 8; ++r) l[r] += ts[r];

    // spill P (16x32) to padded LDS to reform an A fragment
#pragma unroll
    for (int r = 0; r < 8; ++r) {
      pw[(r + 8 * hh) * PLD + colk]      = (f16)p0[r];
      pw[(r + 8 * hh) * PLD + 16 + colk] = (f16)p1[r];
    }
    __syncthreads();

    // O += P * V   (K-dim = 32 keys, 4 N-chunks of 16 over Dh)
    v16h pa  = load_a_frag(pw, PLD, lane);
    v16h vb0 = load_b_frag(&vt_lds[0],  KLD, lane);
    v16h vb1 = load_b_frag(&vt_lds[16], KLD, lane);
    v16h vb2 = load_b_frag(&vt_lds[32], KLD, lane);
    v16h vb3 = load_b_frag(&vt_lds[48], KLD, lane);
    o0 = WMMA_F32_F16(pa, vb0, o0);
    o1 = WMMA_F32_F16(pa, vb1, o1);
    o2 = WMMA_F32_F16(pa, vb2, o2);
    o3 = WMMA_F32_F16(pa, vb3, o3);
  }

  // normalize and store y (f16, [B*T, 1024])
#pragma unroll
  for (int r = 0; r < 8; ++r) {
    int qrow = q0 + r + 8 * hh;
    float inv = 1.0f / l[r];
    size_t base = (size_t)(b * T + qrow) * 1024 + head * DH;
    y[base +  0 + colk] = (f16)(o0[r] * inv);
    y[base + 16 + colk] = (f16)(o1[r] * inv);
    y[base + 32 + colk] = (f16)(o2[r] * inv);
    y[base + 48 + colk] = (f16)(o3[r] * inv);
  }
}

// ---------------------------------------------------------------------------
extern "C" void kernel_launch(void* const* d_in, const int* in_sizes, int n_in,
                              void* d_out, int out_size, void* d_ws, size_t ws_size,
                              hipStream_t stream) {
  const float* x     = (const float*)d_in[0];  // [4,2048,1024]
  const float* Wqkv  = (const float*)d_in[1];  // [1024,3072]
  const float* bqkv  = (const float*)d_in[2];  // [3072]
  const float* Wproj = (const float*)d_in[3];  // [1024,1024]
  const float* bproj = (const float*)d_in[4];  // [1024]

  constexpr int BT = 4 * 2048;  // 8192 rows
  f16* qkv_h = (f16*)d_ws;                       // 8192*3072 f16 = 48 MB
  f16* y_h   = qkv_h + (size_t)BT * 3072;        // 8192*1024 f16 = 16 MB

  // 1) QKV projection: qkv_h = x @ Wqkv + bqkv   (f16 out)
  dim3 g1(3072 / 128, BT / 128);
  gemm_bias_wmma<float, f16><<<g1, 256, 0, stream>>>(x, Wqkv, bqkv, qkv_h,
                                                     BT, 3072, 1024);

  // 2) causal flash attention: y_h = attn(qkv_h)
  dim3 g2(2048 / 128, 4 * 16);
  attn_wmma<<<g2, 256, 0, stream>>>(qkv_h, y_h);

  // 3) output projection: out = y_h @ Wproj + bproj  (fp32 out)
  dim3 g3(1024 / 128, BT / 128);
  gemm_bias_wmma<f16, float><<<g3, 256, 0, stream>>>(y_h, Wproj, bproj,
                                                     (float*)d_out,
                                                     BT, 1024, 1024);
}
